// DynamicWeightProjection_74998718923388
// MI455X (gfx1250) — compile-verified, matrix operand order checked by reading, not run on hardware
//
#include <hip/hip_runtime.h>
#include <hip/hip_bf16.h>

// ---------------------------------------------------------------------------
// MI455X (gfx1250) fused DynamicWeightProjection
//   Kernel A: H = gelu(X @ dw1) -> bf16 ws ; ddv = tanh(X @ dd) -> out
//   Kernel B: w = H_c @ qkw_c ; RMSNorm(half) ; scatter -> out
// Precision: bf16 WMMA (v_wmma_f32_16x16x32_bf16), fp32 accumulate.
// ---------------------------------------------------------------------------

typedef __attribute__((ext_vector_type(16))) __bf16 v16bf;
typedef __attribute__((ext_vector_type(2)))  __bf16 v2bf;
typedef __attribute__((ext_vector_type(8)))  float  v8f;

union FragU { v16bf v; unsigned u[8]; };

// fp32 x2 -> packed bf16 dword (lo = first arg), round-to-nearest-even
#if __has_builtin(__builtin_amdgcn_cvt_pk_bf16_f32)
__device__ __forceinline__ unsigned pack2bf(float a, float b) {
  union { v2bf v; unsigned u; } cv;
  cv.v = __builtin_amdgcn_cvt_pk_bf16_f32(a, b);
  return cv.u;
}
#else
__device__ __forceinline__ unsigned pack2bf(float a, float b) {
  unsigned ua = __float_as_uint(a), ub = __float_as_uint(b);
  ua += 0x7fffu + ((ua >> 16) & 1u);
  ub += 0x7fffu + ((ub >> 16) & 1u);
  return (ua >> 16) | (ub & 0xffff0000u);
}
#endif

// Load one 16x32 bf16 WMMA operand fragment from LDS stored as K-pairs
// (dword = two consecutive-K bf16) at lds[rowoff + pbase + pairIndex].
// ISA layout: lanes 0-15 hold K-pairs {0..3, 8..11}, lanes 16-31 {4..7, 12..15}.
__device__ __forceinline__ v16bf load_frag16(const unsigned* lds, int rowoff,
                                             int pbase, int phalf) {
  FragU f;
#pragma unroll
  for (int v = 0; v < 4; ++v) f.u[v] = lds[rowoff + pbase + phalf + v];
#pragma unroll
  for (int v = 0; v < 4; ++v) f.u[4 + v] = lds[rowoff + pbase + phalf + 8 + v];
  return f.v;
}

#define TOTROWS 8192
#define DDIM    4096
#define N1      512      // dw1 columns (G*C*K)
#define NTOT    640      // + 128 dd columns
#define BM      64       // rows per block
#define KC      32       // K chunk (one WMMA K step)

// flat offsets of the 12 outputs in d_out (fp32 elements)
#define O_PRE_QW1   0u
#define O_PRE_QW2   524288u
#define O_PRE_QDD   2097152u

__global__ __launch_bounds__(256)
void dwp_gemm1(const float* __restrict__ X,      // [8192][4096]
               const float* __restrict__ dw1,    // [4096][512]
               const float* __restrict__ dd,     // [4096][128]
               unsigned short* __restrict__ hw,  // [8192][512] bf16 workspace
               float* __restrict__ out) {
  __shared__ unsigned ldsX[BM * 17];     // 64 rows x 16 K-pairs (+1 pad)
  __shared__ unsigned ldsW[NTOT * 17];   // 640 cols x 16 K-pairs (+1 pad)

  const int tid   = threadIdx.x;
  const int lane  = tid & 31;
  const int w     = tid >> 5;      // 8 waves
  const int wn    = w & 3;         // 4 groups along N (10 tiles each)
  const int wm    = w >> 2;        // 2 groups along M (2 tiles each)
  const int l15   = lane & 15;
  const int phalf = (lane >= 16) ? 4 : 0;
  const int row0  = blockIdx.x * BM;

  v8f acc[2][10] = {};

  const int xr = tid >> 4;   // X staging row 0..15
  const int xp = tid & 15;   // X staging pair 0..15
  const int wp  = tid >> 4;  // W staging pair 0..15 (k = kb + 2*wp)
  const int wnn = tid & 15;  // W staging col base

  for (int kb = 0; kb < DDIM; kb += KC) {
    // ---- stage X tile (64 x 32 fp32 -> bf16 pairs) --------------------
#pragma unroll
    for (int it = 0; it < 4; ++it) {
      int row = xr + it * 16;
      const float2 f2 =
          *(const float2*)(X + (size_t)(row0 + row) * DDIM + kb + 2 * xp);
      ldsX[row * 17 + xp] = pack2bf(f2.x, f2.y);
    }
    if (kb + KC < DDIM)  // hint: global_prefetch_b8 for next X chunk
      __builtin_prefetch(X + (size_t)(row0 + xr) * DDIM + kb + KC, 0, 1);
    // ---- stage W tile (32 x 640, transposed to col-major pairs) -------
    {
      const int k0 = kb + 2 * wp;
      const float* r0 = dw1 + (size_t)k0 * N1;
      const float* r1 = r0 + N1;
#pragma unroll 4
      for (int j = 0; j < 32; ++j) {               // dw1 columns 0..511
        int n = wnn + 16 * j;
        ldsW[n * 17 + wp] = pack2bf(r0[n], r1[n]);
      }
      const float* d0 = dd + (size_t)k0 * 128;
      const float* d1 = d0 + 128;
#pragma unroll
      for (int j = 0; j < 8; ++j) {                // dd columns 0..127
        int n = wnn + 16 * j;
        ldsW[(N1 + n) * 17 + wp] = pack2bf(d0[n], d1[n]);
      }
    }
    __syncthreads();

    // ---- WMMA: 2 M-tiles x 10 N-tiles per wave, double-buffered B -----
    v16bf a0 = load_frag16(ldsX, ((wm * 2 + 0) * 16 + l15) * 17, 0, phalf);
    v16bf a1 = load_frag16(ldsX, ((wm * 2 + 1) * 16 + l15) * 17, 0, phalf);
    v16bf b  = load_frag16(ldsW, ((wn * 10) * 16 + l15) * 17, 0, phalf);
#pragma unroll
    for (int j = 0; j < 10; ++j) {
      v16bf bn = b;
      if (j < 9)
        bn = load_frag16(ldsW, ((wn * 10 + j + 1) * 16 + l15) * 17, 0, phalf);
      acc[0][j] = __builtin_amdgcn_wmma_f32_16x16x32_bf16(
          false, a0, false, b, (short)0, acc[0][j], false, false);
      acc[1][j] = __builtin_amdgcn_wmma_f32_16x16x32_bf16(
          false, a1, false, b, (short)0, acc[1][j], false, false);
      b = bn;
    }
    __syncthreads();
  }

  // ---- epilogue: GELU(erf) -> bf16 ws ; tanh -> dd outputs -------------
  const int rhalf = (lane >= 16) ? 8 : 0;
#pragma unroll
  for (int mi = 0; mi < 2; ++mi) {
#pragma unroll
    for (int j = 0; j < 10; ++j) {
      int col = (wn * 10 + j) * 16 + l15;
#pragma unroll
      for (int v = 0; v < 8; ++v) {
        int row = row0 + (wm * 2 + mi) * 16 + rhalf + v;
        float f = acc[mi][j][v];
        if (col < N1) {
          float g = 0.5f * f * (1.0f + erff(f * 0.70710678118654752f));
          hw[(size_t)row * N1 + col] = (unsigned short)pack2bf(g, g);
        } else {
          int q = col - N1;                      // 0..127
          size_t base = (size_t)O_PRE_QDD + (size_t)((q >> 5) & 1) * 262144u +
                        (size_t)(q >> 6) * 2621440u;
          out[base + (size_t)row * 32 + (q & 31)] = tanhf(f);
        }
      }
    }
  }
}

__global__ __launch_bounds__(256)
void dwp_gemm2(const unsigned short* __restrict__ hw,  // [8192][512] bf16
               const float* __restrict__ qkw,          // [4][128][128]
               float* __restrict__ out) {
  // ldsA: 64 rows x 64 pairs (stride 65); ldsB: 128 cols x 64 pairs (stride 65)
  // eplds (epilogue, 64x128 fp32) aliases the staging space after a barrier.
  __shared__ unsigned lds[64 * 65 + 128 * 65];
  unsigned* ldsA = lds;
  unsigned* ldsB = lds + 64 * 65;
  float*    eplds = (float*)lds;

  const int tid   = threadIdx.x;
  const int lane  = tid & 31;
  const int w     = tid >> 5;
  const int mt    = w & 3;        // M tile 0..3
  const int ng    = w >> 2;       // N half 0..1 (4 N-tiles each)
  const int l15   = lane & 15;
  const int phalf = (lane >= 16) ? 4 : 0;
  const int rhalf = (lane >= 16) ? 8 : 0;
  const int row0  = blockIdx.x * 64;

  const unsigned* hw32 = (const unsigned*)hw;  // bf16-pair view: [row][256]

  for (int c = 0; c < 4; ++c) {
    __syncthreads();  // previous epilogue finished reading aliased LDS
    // ---- stage A: H_c tile via async global->LDS dword copies ---------
    {
      int p   = tid & 63;
      int row = tid >> 6;   // 0..3, step 4
#pragma unroll
      for (int it = 0; it < 16; ++it) {
        int r = row + it * 4;
        unsigned long long ga =
            (unsigned long long)(size_t)(hw32 + (size_t)(row0 + r) * 256 +
                                         c * 64 + p);
        unsigned ldsoff = (unsigned)(size_t)(
            (__attribute__((address_space(3))) const unsigned*)&ldsA[r * 65 + p]);
        asm volatile("global_load_async_to_lds_b32 %0, %1, off"
                     :: "v"(ldsoff), "v"(ga) : "memory");
      }
    }
    // ---- stage B: qkw_c (128x128 fp32) -> col-major bf16 pairs --------
    for (int idx = tid; idx < 128 * 64; idx += 256) {
      int j = idx & 127;
      int p = idx >> 7;
      int k = c * 128 + 2 * p;
      ldsB[j * 65 + p] =
          pack2bf(qkw[(size_t)k * 128 + j], qkw[(size_t)(k + 1) * 128 + j]);
    }
    asm volatile("s_wait_asynccnt 0x0" ::: "memory");
    __syncthreads();

    v8f acc[4] = {};
#pragma unroll
    for (int ks = 0; ks < 4; ++ks) {   // K = 128 -> 4 WMMA steps
      v16bf af = load_frag16(ldsA, (mt * 16 + l15) * 65, ks * 16, phalf);
#pragma unroll
      for (int j = 0; j < 4; ++j) {
        int n = (ng * 4 + j) * 16 + l15;
        v16bf bfr = load_frag16(ldsB, n * 65, ks * 16, phalf);
        acc[j] = __builtin_amdgcn_wmma_f32_16x16x32_bf16(
            false, af, false, bfr, (short)0, acc[j], false, false);
      }
    }
    __syncthreads();  // done with ldsA/ldsB; reuse as eplds
    // ---- dump accumulators to LDS for cross-lane RMSNorm --------------
#pragma unroll
    for (int j = 0; j < 4; ++j) {
      int col = (ng * 4 + j) * 16 + l15;
#pragma unroll
      for (int v = 0; v < 8; ++v) {
        int rl = mt * 16 + rhalf + v;
        eplds[rl * 128 + col] = acc[j][v];
      }
    }
    __syncthreads();
    // ---- RMSNorm (i<2) + scatter: one (row, i) group per thread -------
    {
      int rl = tid >> 2;   // 0..63
      int i  = tid & 3;    // 0..3
      float vals[32];
      float ss = 0.f;
#pragma unroll
      for (int m = 0; m < 32; ++m) {
        vals[m] = eplds[rl * 128 + i * 32 + m];
        ss += vals[m] * vals[m];
      }
      float scale = 1.0f;
      if (i < 2) scale = rsqrtf(ss * (1.0f / 32.0f) + 1e-6f);
      // W1BASE[c] = c*1048576 + (c>>1)*524288 ; W2 = +524288
      size_t base = (size_t)c * 1048576u + (size_t)(c >> 1) * 524288u +
                    (i >= 2 ? (size_t)O_PRE_QW2 : 0u);
      size_t rr = (size_t)(row0 + rl);
      size_t o = base + (rr * 2 + (size_t)(i & 1)) * 32;
#pragma unroll
      for (int m = 0; m < 32; ++m) out[o + m] = vals[m] * scale;
    }
  }
}

extern "C" void kernel_launch(void* const* d_in, const int* in_sizes, int n_in,
                              void* d_out, int out_size, void* d_ws, size_t ws_size,
                              hipStream_t stream) {
  (void)in_sizes; (void)n_in; (void)out_size; (void)ws_size;
  const float* X   = (const float*)d_in[0];   // query_vec [2,4096,4096]
  // d_in[1] = key_vec: unused by the reference
  const float* dw1 = (const float*)d_in[2];   // [4096,1,4,128] == [4096,512]
  const float* qkw = (const float*)d_in[3];   // [1,4,128,4,32] == [4,128,128]
  const float* dd  = (const float*)d_in[4];   // [4096,1,128]   == [4096,128]
  float* out = (float*)d_out;
  unsigned short* hw = (unsigned short*)d_ws; // 8192*512 bf16 = 8 MB

  dwp_gemm1<<<TOTROWS / BM, 256, 0, stream>>>(X, dw1, dd, hw, out);
  dwp_gemm2<<<TOTROWS / 64, 256, 0, stream>>>(hw, qkw, out);
}
